// DAGConcat_60361470378542
// MI455X (gfx1250) — matrix operand order repaired and checked by reading, not compiled, not run on hardware
//
#include <hip/hip_runtime.h>
#include <hip/hip_bf16.h>
#include <math.h>

// ---------------------------------------------------------------------------
// MI455X (gfx1250, wave32) DAG-GNN forward pass.
// Dense math: v_wmma_f32_16x16x32_f16, register-blocked MT row-tiles per wave
// (one B fragment feeds MT back-to-back WMMAs), double-buffered LDS staging of
// the f32->f16 A tile, k-loop unrolled x2 with ping-pong B registers (no
// copies), fragment ds_loads hoisted ahead of the staging stores so partial
// s_wait_dscnt suffices. Weights converted to f16 once per launch (~25MB ->
// resident in 192MB L2 across the 2x128-step GNN scan).
// ---------------------------------------------------------------------------

typedef __attribute__((ext_vector_type(16))) _Float16 v16h;
typedef __attribute__((ext_vector_type(4)))  _Float16 v4h;
typedef __attribute__((ext_vector_type(8)))  float    v8f;
typedef __attribute__((ext_vector_type(8)))  unsigned v8u;

#define BATCH 32
#define NNODE 128
#define EMB   1024
#define HID   512

__device__ __forceinline__ float sigmoidf_(float x) { return 1.0f / (1.0f + __expf(-x)); }

// ---------------------------------------------------------------------------
// C[M x N] = act( A[M x K](f32, row stride lda) @ Bh[N x K](f16 row-major)^T + bias )
// grid = (N/64, M/(16*MT)), block = 128 = 4 waves. Wave w owns col tile 16*w
// and MT row tiles. K/32 must be even (all callers: 16/32/96 chunks).
// ---------------------------------------------------------------------------
template <int MT>
__global__ __launch_bounds__(128)
void wmma_gemm_kernel(const float* __restrict__ A, int lda,
                      const _Float16* __restrict__ Bh,
                      const float* __restrict__ bias,
                      float* __restrict__ C, int ldc,
                      int K, int act)
{
  __shared__ unsigned As32[2 * MT * 256];        // 2 bufs x MT tiles x (16x32 f16)
  const int tid   = threadIdx.x;
  const int wave  = tid >> 5;
  const int lane  = tid & 31;
  const int khalf = lane >> 4;                   // 0: K 0..7,16..23; 1: K 8..15,24..31 of chunk
  const int nloc  = lane & 15;
  const int rowBase = blockIdx.y * (16 * MT);
  const int col     = (blockIdx.x << 6) + (wave << 4) + nloc;

  // A staging: 128 thr * 4 f32 per tile = 16x32
  const int am = tid >> 3;                       // row 0..15 within tile
  const int ak = (tid & 7) << 2;                 // k offset 0,4,..,28

  const unsigned* __restrict__ Bu = (const unsigned*)(Bh + (size_t)col * K);
  const int apair = nloc * 16;                   // LDS dword row base for A frag

  v8f acc[MT];
#pragma unroll
  for (int t = 0; t < MT; ++t) acc[t] = (v8f){};

  // B fragment: lane(0-15)=N, VGPR v<4 -> K = 8*khalf + 2v ; v>=4 -> 16+8*khalf+2(v-4) (packed pairs)
  auto loadB = [&](int k0) -> v8u {
    v8u bu;
    const unsigned* bp = Bu + ((k0 >> 1) + (khalf << 2));
#pragma unroll
    for (int v = 0; v < 8; ++v) bu[v] = bp[(v & 3) + ((v >> 2) << 3)];
    return bu;
  };

  // Stage A tile: all MT global loads first (single clause/wait), then cvt+store.
  auto stage = [&](int k0, int buf) {
    float4 f[MT];
#pragma unroll
    for (int t = 0; t < MT; ++t)
      f[t] = *(const float4*)(A + (size_t)(rowBase + t * 16 + am) * lda + (k0 + ak));
#pragma unroll
    for (int t = 0; t < MT; ++t) {
      v4h h; h[0] = (_Float16)f[t].x; h[1] = (_Float16)f[t].y;
             h[2] = (_Float16)f[t].z; h[3] = (_Float16)f[t].w;
      *(uint2*)&As32[buf * (MT * 256) + t * 256 + tid * 2] =
          __builtin_bit_cast(uint2, h);          // ds_store_b64
    }
  };

  // A fragments: lane(0-15)=M, VGPR j<4 -> K = 8*khalf+2j ; j>=4 -> 16+8*khalf+2(j-4)
  auto loadFrags = [&](int buf, v8u (&au)[MT]) {
    const unsigned* base = &As32[buf * (MT * 256)];
#pragma unroll
    for (int t = 0; t < MT; ++t) {
#pragma unroll
      for (int j = 0; j < 4; ++j) {
        au[t][j]     = base[t * 256 + apair + (khalf << 2) + j];
        au[t][4 + j] = base[t * 256 + apair + 8 + (khalf << 2) + j];
      }
    }
  };

  auto mma = [&](v8u (&au)[MT], v8u b) {
    v16h bv = __builtin_bit_cast(v16h, b);
#pragma unroll
    for (int t = 0; t < MT; ++t) {
      v16h av = __builtin_bit_cast(v16h, au[t]);
      acc[t] = __builtin_amdgcn_wmma_f32_16x16x32_f16(false, av, false, bv,
                                                      (short)0, acc[t], false, false);
    }
  };

  stage(0, 0);
  v8u b0 = loadB(0);
  __syncthreads();

  for (int k0 = 0; k0 < K; k0 += 64) {           // 2 chunks per iteration
    v8u au[MT];

    // ---- chunk k0: buf0 + b0; stage k0+32 into buf1, load b1 --------------
    loadFrags(0, au);                            // ds_loads first...
    stage(k0 + 32, 1);                           // ...then ds_stores (partial wait ok)
    v8u b1 = loadB(k0 + 32);
    if (k0 + 96 < K) __builtin_prefetch(Bu + ((k0 + 96) >> 1), 0, 1);
    mma(au, b0);
    __syncthreads();

    // ---- chunk k0+32: buf1 + b1; stage k0+64 into buf0, load next b0 ------
    loadFrags(1, au);
    if (k0 + 64 < K) {
      stage(k0 + 64, 0);
      b0 = loadB(k0 + 64);
      if (k0 + 128 < K) __builtin_prefetch(Bu + ((k0 + 128) >> 1), 0, 1);
    }
    mma(au, b1);
    __syncthreads();
  }

  float bvs = bias ? bias[col] : 0.0f;
#pragma unroll
  for (int t = 0; t < MT; ++t) {
#pragma unroll
    for (int v = 0; v < 8; ++v) {                // C/D: M = v + 8*khalf, N = lane&15
      int m = rowBase + t * 16 + v + (khalf << 3);
      float val = acc[t][v] + bvs;
      if (act) val = fmaxf(val, 0.0f);
      C[(size_t)m * ldc + col] = val;
    }
  }
}

// ---------------------------------------------------------------------------
__global__ void cvt_kernel(const float* __restrict__ s, _Float16* __restrict__ d, int n) {
  int i = blockIdx.x * 256 + threadIdx.x;
  if (i < n) d[i] = (_Float16)s[i];
}

__global__ void zero_kernel(float* p, int n) {
  int i = blockIdx.x * 256 + threadIdx.x;
  if (i < n) p[i] = 0.0f;
}

// LSTM cell: g = Gi[t] + Gh (biases already folded in by the GEMMs)
__global__ void lstm_cell_kernel(const float* __restrict__ Gi, const float* __restrict__ Gh,
                                 float* __restrict__ h, float* __restrict__ c,
                                 float* __restrict__ ys)
{
  int idx = blockIdx.x * 256 + threadIdx.x;      // 128*512
  int row = idx >> 9, k = idx & 511;
  const float* gi = Gi + (size_t)row * 2048;
  const float* gh = Gh + (size_t)row * 2048;
  float i_ = gi[k]        + gh[k];
  float f_ = gi[512 + k]  + gh[512 + k];
  float g_ = gi[1024 + k] + gh[1024 + k];
  float o_ = gi[1536 + k] + gh[1536 + k];
  float cv = sigmoidf_(f_) * c[idx] + sigmoidf_(i_) * tanhf(g_);
  float hv = sigmoidf_(o_) * tanhf(cv);
  c[idx] = cv; h[idx] = hv; ys[idx] = hv;
}

// h = gruC(q, M) + gruP(M, q).  Gq = q@[cWih;pWhh]^T, Gm = M@[cWhh;pWih]^T (no bias yet)
__global__ void gru_combine_kernel(const float* __restrict__ Hprev, int i,
                                   const float* __restrict__ Mb,
                                   const float* __restrict__ Gq, const float* __restrict__ Gm,
                                   const float* __restrict__ cbih, const float* __restrict__ cbhh,
                                   const float* __restrict__ pbih, const float* __restrict__ pbhh,
                                   float* __restrict__ H1)
{
  int idx = blockIdx.x * 256 + threadIdx.x;      // 32*512
  int b = idx >> 9, k = idx & 511;
  float q = Hprev[((size_t)b * NNODE + i) * HID + k];
  float m = Mb[idx];
  const float* gq = Gq + (size_t)b * 3072;
  const float* gm = Gm + (size_t)b * 3072;
  // gruC: x=q (gi = Gq[0:1536]), h=m (gh = Gm[0:1536])
  float r  = sigmoidf_((gq[k]        + cbih[k])        + (gm[k]        + cbhh[k]));
  float z  = sigmoidf_((gq[512 + k]  + cbih[512 + k])  + (gm[512 + k]  + cbhh[512 + k]));
  float n  = tanhf((gq[1024 + k] + cbih[1024 + k]) + r * (gm[1024 + k] + cbhh[1024 + k]));
  float hC = (1.0f - z) * n + z * m;
  // gruP: x=m (gi = Gm[1536:3072]), h=q (gh = Gq[1536:3072])
  float r2  = sigmoidf_((gm[1536 + k] + pbih[k])        + (gq[1536 + k] + pbhh[k]));
  float z2  = sigmoidf_((gm[2048 + k] + pbih[512 + k])  + (gq[2048 + k] + pbhh[512 + k]));
  float n2  = tanhf((gm[2560 + k] + pbih[1024 + k]) + r2 * (gq[2560 + k] + pbhh[1024 + k]));
  float hP  = (1.0f - z2) * n2 + z2 * q;
  H1[((size_t)b * NNODE + i) * HID + k] = hC + hP;
}

// Per-step attention: alpha_n = q.wq + H1[n].wk + ab (masked), softmax over n<i,
// M = sum_n w_n * (V0[n]*s + V1[n]*(1-s)).  One block per batch element b.
__global__ __launch_bounds__(256)
void attn_kernel(const float* __restrict__ Hprev, const float* __restrict__ H1,
                 const float* __restrict__ Vcat,
                 const float* __restrict__ adj, const float* __restrict__ smask,
                 const float* __restrict__ aW, const float* __restrict__ abp,
                 int l, int i, float* __restrict__ Mb)
{
  const int b = blockIdx.x;
  const int tid = threadIdx.x;
  __shared__ float sAlpha[NNODE];
  __shared__ float sRed[256];
  __shared__ float sQdot;

  const float ab = abp[l];
  const float* q = Hprev + ((size_t)b * NNODE + i) * HID;

  // q . wq  (wq = aW[0:512])
  float p = q[tid] * aW[tid] + q[tid + 256] * aW[tid + 256];
  sRed[tid] = p; __syncthreads();
  for (int s = 128; s > 0; s >>= 1) { if (tid < s) sRed[tid] += sRed[tid + s]; __syncthreads(); }
  if (tid == 0) sQdot = sRed[0];
  __syncthreads();

  // alpha[n] for n < i, one wave per n (wave32 shuffle reduce)
  const float* wk = aW + 512;
  const int wave = tid >> 5, lane = tid & 31;
  for (int n = wave; n < i; n += 8) {
    const float* hr = H1 + ((size_t)b * NNODE + n) * HID;
    float d = 0.0f;
    for (int k = lane; k < HID; k += 32) d += hr[k] * wk[k];
#pragma unroll
    for (int off = 16; off > 0; off >>= 1) d += __shfl_xor(d, off, 32);
    if (lane == 0) {
      float a = sQdot + d + ab;
      a -= (1.0f - adj[((size_t)b * NNODE + i) * NNODE + n]) * 1e30f;
      sAlpha[n] = a;
    }
  }
  __syncthreads();

  // softmax over n < i
  float mv = -INFINITY;
  for (int n = tid; n < i; n += 256) mv = fmaxf(mv, sAlpha[n]);
  sRed[tid] = mv; __syncthreads();
  for (int s = 128; s > 0; s >>= 1) { if (tid < s) sRed[tid] = fmaxf(sRed[tid], sRed[tid + s]); __syncthreads(); }
  float maxv = sRed[0]; __syncthreads();
  float sum = 0.0f;
  for (int n = tid; n < i; n += 256) { float e = __expf(sAlpha[n] - maxv); sAlpha[n] = e; sum += e; }
  sRed[tid] = sum; __syncthreads();
  for (int s = 128; s > 0; s >>= 1) { if (tid < s) sRed[tid] += sRed[tid + s]; __syncthreads(); }
  float inv = 1.0f / sRed[0];

  // M[h] = sum_n w_n * (V0*s + V1*(1-s));  Vcat[b][n] = [V0(512) | V1(512)]
  for (int h = tid; h < HID; h += 256) {
    float acc = 0.0f;
    for (int n = 0; n < i; ++n) {
      float w = sAlpha[n] * inv;
      float s = smask[((size_t)b * NNODE + i) * NNODE + n];
      const float* vr = Vcat + ((size_t)b * NNODE + n) * 1024;
      acc += w * (vr[h] * s + vr[512 + h] * (1.0f - s));
    }
    Mb[b * HID + h] = acc;
  }
}

// Hcat = [H0 | H1_l1 | H1_l2 | features | lstm_out] along dim2 -> (4096, 3072)
__global__ void concat_kernel(const float* __restrict__ H0, const float* __restrict__ H1a,
                              const float* __restrict__ H1b, const float* __restrict__ feat,
                              const float* __restrict__ lstm, float* __restrict__ Hcat, int total)
{
  int i = blockIdx.x * 256 + threadIdx.x;
  if (i >= total) return;
  int r = i / 3072, c = i % 3072;
  float v;
  if      (c < 512)  v = H0[(size_t)r * 512 + c];
  else if (c < 1024) v = H1a[(size_t)r * 512 + (c - 512)];
  else if (c < 1536) v = H1b[(size_t)r * 512 + (c - 1024)];
  else if (c < 2560) v = feat[(size_t)r * 1024 + (c - 1536)];
  else               v = lstm[(size_t)r * 512 + (c - 2560)];
  Hcat[i] = v;
}

// Final 7-class head (N=7 not WMMA-tileable; tiny: 4096x512x7)
__global__ void out_kernel(const float* __restrict__ X, const float* __restrict__ W,
                           const float* __restrict__ b, float* __restrict__ out, int rows)
{
  int i = blockIdx.x * 256 + threadIdx.x;
  if (i >= rows * 7) return;
  int r = i / 7, c = i % 7;
  const float* x = X + (size_t)r * 512;
  const float* w = W + (size_t)c * 512;
  float acc = b[c];
  for (int k = 0; k < 512; ++k) acc += x[k] * w[k];
  out[i] = acc;
}

// ---------------------------------------------------------------------------
static inline char* wsalloc(char*& p, size_t bytes) {
  char* r = p;
  p += (bytes + 255) & ~(size_t)255;
  return r;
}

extern "C" void kernel_launch(void* const* d_in, const int* in_sizes, int n_in,
                              void* d_out, int out_size, void* d_ws, size_t ws_size,
                              hipStream_t stream)
{
  (void)in_sizes; (void)n_in; (void)out_size; (void)ws_size;

  const float* features = (const float*)d_in[0];
  const float* adj      = (const float*)d_in[1];
  const float* smask    = (const float*)d_in[2];
  const float* lstm_Wih = (const float*)d_in[5];
  const float* lstm_Whh = (const float*)d_in[6];
  const float* lstm_bih = (const float*)d_in[7];
  const float* lstm_bhh = (const float*)d_in[8];
  const float* fc1_W    = (const float*)d_in[9];
  const float* fc1_b    = (const float*)d_in[10];
  const float* attn_W   = (const float*)d_in[11];
  const float* attn_b   = (const float*)d_in[12];
  const float* Wr0      = (const float*)d_in[13];
  const float* Wr1      = (const float*)d_in[14];
  const float* gC_Wih   = (const float*)d_in[15];
  const float* gC_Whh   = (const float*)d_in[16];
  const float* gC_bih   = (const float*)d_in[17];
  const float* gC_bhh   = (const float*)d_in[18];
  const float* gP_Wih   = (const float*)d_in[19];
  const float* gP_Whh   = (const float*)d_in[20];
  const float* gP_bih   = (const float*)d_in[21];
  const float* gP_bhh   = (const float*)d_in[22];
  const float* mlp0_W   = (const float*)d_in[23];
  const float* mlp0_b   = (const float*)d_in[24];
  const float* mlp1_W   = (const float*)d_in[25];
  const float* mlp1_b   = (const float*)d_in[26];
  const float* out_W    = (const float*)d_in[27];
  const float* out_b    = (const float*)d_in[28];

  // ---- workspace carve-out -------------------------------------------------
  char* p = (char*)d_ws;
  _Float16* WihH   = (_Float16*)wsalloc(p, (size_t)2048 * 1024 * 2);
  _Float16* WhhH   = (_Float16*)wsalloc(p, (size_t)2048 * 512 * 2);
  _Float16* fc1H   = (_Float16*)wsalloc(p, (size_t)512 * 1024 * 2);
  _Float16* WqCat[2], *WmCat[2], *WrCat[2];
  for (int l = 0; l < 2; ++l) {
    WqCat[l] = (_Float16*)wsalloc(p, (size_t)3072 * 512 * 2);
    WmCat[l] = (_Float16*)wsalloc(p, (size_t)3072 * 512 * 2);
    WrCat[l] = (_Float16*)wsalloc(p, (size_t)1024 * 512 * 2);
  }
  _Float16* mlp0H = (_Float16*)wsalloc(p, (size_t)512 * 3072 * 2);
  _Float16* mlp1H = (_Float16*)wsalloc(p, (size_t)512 * 512 * 2);

  float* GiAll   = (float*)wsalloc(p, (size_t)4096 * 2048 * 4);
  float* Ghbuf   = (float*)wsalloc(p, (size_t)128 * 2048 * 4);
  float* hbuf    = (float*)wsalloc(p, (size_t)128 * 512 * 4);
  float* cbuf    = (float*)wsalloc(p, (size_t)128 * 512 * 4);
  float* lstmOut = (float*)wsalloc(p, (size_t)4096 * 512 * 4);
  float* H0      = (float*)wsalloc(p, (size_t)4096 * 512 * 4);
  float* H1a     = (float*)wsalloc(p, (size_t)4096 * 512 * 4);
  float* H1b     = (float*)wsalloc(p, (size_t)4096 * 512 * 4);
  float* Vcat    = (float*)wsalloc(p, (size_t)4096 * 1024 * 4);
  float* Mbuf    = (float*)wsalloc(p, (size_t)32 * 512 * 4);
  float* Gq      = (float*)wsalloc(p, (size_t)32 * 3072 * 4);
  float* Gm      = (float*)wsalloc(p, (size_t)32 * 3072 * 4);
  float* Hcat    = (float*)wsalloc(p, (size_t)4096 * 3072 * 4);
  float* X1      = (float*)wsalloc(p, (size_t)4096 * 512 * 4);
  float* X2      = (float*)wsalloc(p, (size_t)4096 * 512 * 4);

  auto gemm = [&](const float* A, int lda, const _Float16* Bh, const float* bias,
                  float* C, int ldc, int M, int N, int K, int act) {
    if (M % 64 == 0) {
      dim3 grid(N / 64, M / 64);
      wmma_gemm_kernel<4><<<grid, 128, 0, stream>>>(A, lda, Bh, bias, C, ldc, K, act);
    } else {            // M == 32 (scan-step GEMMs)
      dim3 grid(N / 64, M / 32);
      wmma_gemm_kernel<2><<<grid, 128, 0, stream>>>(A, lda, Bh, bias, C, ldc, K, act);
    }
  };
  auto zero = [&](float* q, int n) { zero_kernel<<<(n + 255) / 256, 256, 0, stream>>>(q, n); };
  auto cvt  = [&](const float* s, _Float16* d, int n) {
    cvt_kernel<<<(n + 255) / 256, 256, 0, stream>>>(s, d, n);
  };

  // ---- weight conversion (once per launch; deterministic) ------------------
  cvt(lstm_Wih, WihH, 2048 * 1024);
  cvt(lstm_Whh, WhhH, 2048 * 512);
  cvt(fc1_W,    fc1H, 512 * 1024);
  for (int l = 0; l < 2; ++l) {
    const size_t wo = (size_t)l * 1536 * 512;
    cvt(gC_Wih + wo, WqCat[l],                1536 * 512);   // q @ cWih^T
    cvt(gP_Whh + wo, WqCat[l] + 1536 * 512,   1536 * 512);   // q @ pWhh^T
    cvt(gC_Whh + wo, WmCat[l],                1536 * 512);   // M @ cWhh^T
    cvt(gP_Wih + wo, WmCat[l] + 1536 * 512,   1536 * 512);   // M @ pWih^T
    cvt(Wr0 + (size_t)l * 512 * 512, WrCat[l],             512 * 512);
    cvt(Wr1 + (size_t)l * 512 * 512, WrCat[l] + 512 * 512, 512 * 512);
  }
  cvt(mlp0_W, mlp0H, 512 * 3072);
  cvt(mlp1_W, mlp1H, 512 * 512);

  // ---- LSTM over axis 0 (T=32, batch=128) ----------------------------------
  gemm(features, EMB, WihH, lstm_bih, GiAll, 2048, 4096, 2048, 1024, 0);
  zero(hbuf, 128 * 512); zero(cbuf, 128 * 512);
  for (int t = 0; t < 32; ++t) {
    gemm(hbuf, 512, WhhH, lstm_bhh, Ghbuf, 2048, 128, 2048, 512, 0);
    lstm_cell_kernel<<<(128 * 512) / 256, 256, 0, stream>>>(
        GiAll + (size_t)t * 128 * 2048, Ghbuf, hbuf, cbuf,
        lstmOut + (size_t)t * 128 * 512);
  }

  // ---- H0 = relu(features @ fc1^T + b) -------------------------------------
  gemm(features, EMB, fc1H, fc1_b, H0, 512, 4096, 512, 1024, 1);

  // ---- GNN layers: sequential scan over nodes ------------------------------
  const float* Hl = H0;
  float* H1s[2] = {H1a, H1b};
  for (int l = 0; l < 2; ++l) {
    float* H1 = H1s[l];
    zero(Mbuf, 32 * 512);                        // step 0 has M = 0
    for (int i = 0; i < NNODE; ++i) {
      if (i > 0)
        attn_kernel<<<BATCH, 256, 0, stream>>>(Hl, H1, Vcat, adj, smask,
                                               attn_W + (size_t)l * 1024, attn_b,
                                               l, i, Mbuf);
      gemm(Hl + (size_t)i * HID, NNODE * HID, WqCat[l], nullptr, Gq, 3072, 32, 3072, 512, 0);
      gemm(Mbuf, 512,                          WmCat[l], nullptr, Gm, 3072, 32, 3072, 512, 0);
      gru_combine_kernel<<<(32 * 512) / 256, 256, 0, stream>>>(
          Hl, i, Mbuf, Gq, Gm,
          gC_bih + (size_t)l * 1536, gC_bhh + (size_t)l * 1536,
          gP_bih + (size_t)l * 1536, gP_bhh + (size_t)l * 1536, H1);
      gemm(H1 + (size_t)i * HID, NNODE * HID, WrCat[l], nullptr,
           Vcat + (size_t)i * 1024, NNODE * 1024, 32, 1024, 512, 0);
    }
    Hl = H1;
  }

  // ---- concat + MLP head ---------------------------------------------------
  {
    int total = 4096 * 3072;
    concat_kernel<<<(total + 255) / 256, 256, 0, stream>>>(H0, H1a, H1b, features,
                                                           lstmOut, Hcat, total);
  }
  gemm(Hcat, 3072, mlp0H, mlp0_b, X1, 512, 4096, 512, 3072, 1);
  gemm(X1,   512,  mlp1H, mlp1_b, X2, 512, 4096, 512, 512,  1);
  out_kernel<<<(4096 * 7 + 255) / 256, 256, 0, stream>>>(X2, out_W, out_b,
                                                         (float*)d_out, 4096);
}
